// SetAbstraction_SEG_79860621902604
// MI455X (gfx1250) — matrix operand order repaired and checked by reading, not compiled, not run on hardware
//
#include <hip/hip_runtime.h>

// ---------------------------------------------------------------------------
// MI455X / gfx1250 implementation.
// wave32, WMMA 16x16x32 bf16 with fp32 accumulate for all large GEMMs.
// Main GEMM: 128x128 block tile, double-buffered LDS, global_prefetch_b8
// look-ahead. Workspace budget ~351 MB (layout below).
// ---------------------------------------------------------------------------

#define BQ    8
#define NPT   4096
#define SN    1024
#define NS    32
#define DIMD  64
#define DIN   67
#define NG    2217      // real G columns
#define NGP   2304      // padded G columns (18 * 128)
#define EPSBN 1e-5f

typedef __attribute__((ext_vector_type(8)))  float          v8f;
typedef __attribute__((ext_vector_type(16))) __bf16         v16bf;
typedef __attribute__((ext_vector_type(8)))  __bf16         v8bf;
typedef __attribute__((ext_vector_type(8)))  unsigned short us8v;

static __device__ inline unsigned short f2bf(float f) {
    unsigned int u = __builtin_bit_cast(unsigned int, f);
    u += 0x7fffu + ((u >> 16) & 1u);          // round-to-nearest-even
    return (unsigned short)(u >> 16);
}

static __device__ inline v16bf mk16(us8v lo, us8v hi) {
    v8bf l = __builtin_bit_cast(v8bf, lo);
    v8bf h = __builtin_bit_cast(v8bf, hi);
    return __builtin_shufflevector(l, h, 0,1,2,3,4,5,6,7,8,9,10,11,12,13,14,15);
}

// A fragment (16x32 bf16): lane holds row M=lane%16; element e -> K = e + 16*(lane>>4)
// => two contiguous 8-half (16B) runs at k0+16h and k0+16h+8.
static __device__ inline v16bf loadA_frag(const unsigned short* row, int k0, int h) {
    us8v lo = *(const us8v*)(row + k0 + 16 * h);
    us8v hi = *(const us8v*)(row + k0 + 16 * h + 8);
    return mk16(lo, hi);
}
// B fragment from LDS stored transposed [n][k] (32-half rows): element e -> K = e + 16*(lane>>4)
static __device__ inline v16bf loadB_fragT(const unsigned short* nrow, int h) {
    us8v lo = *(const us8v*)(nrow + 16 * h);
    us8v hi = *(const us8v*)(nrow + 16 * h + 8);
    return mk16(lo, hi);
}
// B fragment from pre-swizzled global weights: 16 contiguous halves per lane
static __device__ inline v16bf loadB_swz(const unsigned short* p) {
    us8v lo = *(const us8v*)(p);
    us8v hi = *(const us8v*)(p + 8);
    return mk16(lo, hi);
}

static __device__ inline v8f wmma_bf(v16bf a, v16bf b, v8f c) {
    return __builtin_amdgcn_wmma_f32_16x16x32_bf16(false, a, false, b, (short)0, c, false, false);
}
static __device__ inline v8f vzero8() { v8f z; for (int i = 0; i < 8; ++i) z[i] = 0.f; return z; }

// ---------------------------------------------------------------------------
// 0. init: zero cos accumulator + unique flags
// ---------------------------------------------------------------------------
__global__ void k_init(float* cosacc, int* flags) {
    int t = blockIdx.x * 256 + threadIdx.x;
    if (t < 8)    cosacc[t] = 0.f;
    if (t < NPT)  flags[t]  = 0;
}

// ---------------------------------------------------------------------------
// 1. ball query: first NS indices with dist^2 <= r^2 (ascending), pad w/ first
// ---------------------------------------------------------------------------
__global__ __launch_bounds__(256) void k_ballquery(const float* __restrict__ xyz,
                                                   int* __restrict__ gidx) {
    __shared__ float sx[NPT], sy[NPT], sz[NPT];
    const int tid = threadIdx.x, b = blockIdx.y;
    const int s = blockIdx.x * 256 + tid;
    for (int i = tid; i < NPT; i += 256) {
        sx[i] = xyz[((size_t)b * 3 + 0) * NPT + i];
        sy[i] = xyz[((size_t)b * 3 + 1) * NPT + i];
        sz[i] = xyz[((size_t)b * 3 + 2) * NPT + i];
    }
    __syncthreads();
    const float qx = sx[s], qy = sy[s], qz = sz[s];
    int* out = gidx + ((size_t)b * NPT + s) * NS;
    int cnt = 0, first = 0;
    for (int n = 0; n < NPT && cnt < NS; ++n) {
        float dx = qx - sx[n], dy = qy - sy[n], dz = qz - sz[n];
        float d2 = dx * dx + dy * dy + dz * dz;
        if (d2 <= 0.04f) { if (cnt == 0) first = n; out[cnt++] = n; }
    }
    for (int k = cnt; k < NS; ++k) out[k] = first;
}

// ---------------------------------------------------------------------------
// 2. fused MLP1(67->67) + MLP2(67->1024) + BN + noise -> logits[b][s][n]
// ---------------------------------------------------------------------------
__global__ __launch_bounds__(256) void k_logits(
    const float* __restrict__ points, const float* __restrict__ xyz,
    const float* __restrict__ noise,
    const float* __restrict__ w1w, const float* __restrict__ w1b,
    const float* __restrict__ g1, const float* __restrict__ b1,
    const float* __restrict__ w2w, const float* __restrict__ w2b,
    const float* __restrict__ g2, const float* __restrict__ b2,
    float* __restrict__ logits) {
    const int n = blockIdx.x, b = blockIdx.y, tid = threadIdx.x;
    __shared__ float sa[DIN], sh[DIN];
    if (tid < DIMD)                sa[tid] = points[((size_t)b * DIMD + tid) * NPT + n];
    else if (tid < DIN)            sa[tid] = xyz[((size_t)b * 3 + (tid - DIMD)) * NPT + n];
    __syncthreads();
    const float bns = rsqrtf(1.f + EPSBN);
    if (tid < DIN) {
        float acc = w1b[tid];
        const float* wr = w1w + tid * DIN;
        for (int c = 0; c < DIN; ++c) acc += sa[c] * wr[c];
        sh[tid] = acc * (g1[tid] * bns) + b1[tid];
    }
    __syncthreads();
    for (int s = tid; s < SN; s += 256) {
        float acc = w2b[s];
        const float* wr = w2w + s * DIN;
        for (int c = 0; c < DIN; ++c) acc += sh[c] * wr[c];
        float v = acc * (g2[s] * bns) + b2[s];
        v = v + 5.0f * noise[((size_t)b * SN + s) * NPT + n];   // /GAMMA == /1
        logits[((size_t)b * SN + s) * NPT + n] = v;
    }
}

// ---------------------------------------------------------------------------
// 3. softmax rows -> swb (bf16), swbT (bf16 transposed), norm, idx0 (b==0)
// ---------------------------------------------------------------------------
__global__ __launch_bounds__(256) void k_softmax(
    const float* __restrict__ logits, unsigned short* __restrict__ swb,
    unsigned short* __restrict__ swbT, float* __restrict__ norm,
    int* __restrict__ idx0) {
    const int s = blockIdx.x, b = blockIdx.y, tid = threadIdx.x;
    const float* row = logits + ((size_t)b * SN + s) * NPT;
    __shared__ float rm[256], r2[256];
    __shared__ int   ri[256];
    float m = -3.0e38f; int mi = 0;
    for (int n = tid; n < NPT; n += 256) { float v = row[n]; if (v > m) { m = v; mi = n; } }
    rm[tid] = m; ri[tid] = mi; __syncthreads();
    for (int off = 128; off; off >>= 1) {
        if (tid < off) {
            if (rm[tid + off] > rm[tid] ||
                (rm[tid + off] == rm[tid] && ri[tid + off] < ri[tid])) {
                rm[tid] = rm[tid + off]; ri[tid] = ri[tid + off];
            }
        }
        __syncthreads();
    }
    const float rowmax = rm[0]; const int am = ri[0];
    __syncthreads();
    float se = 0.f, se2 = 0.f;
    for (int n = tid; n < NPT; n += 256) {
        float e = __expf(row[n] - rowmax); se += e; se2 += e * e;
    }
    rm[tid] = se; r2[tid] = se2; __syncthreads();
    for (int off = 128; off; off >>= 1) {
        if (tid < off) { rm[tid] += rm[tid + off]; r2[tid] += r2[tid + off]; }
        __syncthreads();
    }
    const float S = rm[0], inv = 1.f / rm[0], S2 = r2[0];
    for (int n = tid; n < NPT; n += 256) {
        unsigned short u = f2bf(__expf(row[n] - rowmax) * inv);
        swb [((size_t)b * SN + s) * NPT + n] = u;
        swbT[((size_t)b * NPT + n) * SN + s] = u;
    }
    if (tid == 0) {
        norm[b * SN + s] = sqrtf(S2) / S;
        if (b == 0) idx0[s] = am;
    }
}

__global__ void k_markflags(const int* __restrict__ idx0, int* __restrict__ flags) {
    int s = blockIdx.x * 256 + threadIdx.x;
    if (s < SN) flags[idx0[s]] = 1;
}

// ---------------------------------------------------------------------------
// 4. generic 128x128 WMMA GEMM, K multiple of 32. A: MxK bf16 rm, B: KxldB bf16 rm
//    Double-buffered LDS (one barrier per K-step) + global_prefetch look-ahead.
//    EPI 0: C[gm*ldc+gn] = acc      EPI 1: cos-loss accumulation
// ---------------------------------------------------------------------------
template <int EPI>
__global__ __launch_bounds__(256) void k_gemm(
    const unsigned short* __restrict__ Abase, const unsigned short* __restrict__ Bbase,
    float* __restrict__ C, int K, int ldb, int ldc,
    size_t aBS, size_t bBS,
    const float* __restrict__ normBase, int normBS, float* __restrict__ cosacc) {
    const int tid = threadIdx.x, bz = blockIdx.z;
    const unsigned short* A = Abase + (size_t)bz * aBS;
    const unsigned short* B = Bbase + (size_t)bz * bBS;
    const int m0 = blockIdx.y * 128, n0 = blockIdx.x * 128;
    __shared__ unsigned short lA[2][128 * 32];   // [m][k]
    __shared__ unsigned short lB[2][128 * 32];   // [n][k] (transposed)
    const int wave = tid >> 5, lane = tid & 31;
    const int wm = wave >> 2, wn = wave & 3;
    const int h = lane >> 4, l16 = lane & 15;
    const int ar = tid >> 1, ac = (tid & 1) * 16;
    const int bk = tid & 31, bn = (tid >> 5) * 16;

    auto load_tile = [&](int k0, int buf) {
        const unsigned short* asrc = A + (size_t)(m0 + ar) * K + k0 + ac;
        *(uint4*)(&lA[buf][ar * 32 + ac])     = *(const uint4*)(asrc);
        *(uint4*)(&lA[buf][ar * 32 + ac + 8]) = *(const uint4*)(asrc + 8);
        const unsigned short* bsrc = B + (size_t)(k0 + bk) * ldb + n0 + bn;
        unsigned short tmp[16];
        *(uint4*)(&tmp[0]) = *(const uint4*)(bsrc);
        *(uint4*)(&tmp[8]) = *(const uint4*)(bsrc + 8);
        #pragma unroll
        for (int j = 0; j < 16; ++j) lB[buf][(bn + j) * 32 + bk] = tmp[j];
    };

    v8f acc[4][2];
    for (int mt = 0; mt < 4; ++mt)
        for (int nt = 0; nt < 2; ++nt) acc[mt][nt] = vzero8();

    const int KT = K >> 5;
    load_tile(0, 0);
    __syncthreads();
    for (int kt = 0; kt < KT; ++kt) {
        const int cur = kt & 1;
        if (kt + 1 < KT) {
            if (kt + 4 < KT) {   // GL2 look-ahead: lowers to global_prefetch_b8
                __builtin_prefetch(A + (size_t)(m0 + ar) * K + (kt + 4) * 32 + ac, 0, 1);
                __builtin_prefetch(B + (size_t)((kt + 4) * 32 + bk) * ldb + n0 + bn, 0, 1);
            }
            load_tile((kt + 1) * 32, cur ^ 1);
        }
        v16bf af[4], bf[2];
        #pragma unroll
        for (int mt = 0; mt < 4; ++mt)
            af[mt] = loadA_frag(&lA[cur][(wm * 64 + mt * 16 + l16) * 32], 0, h);
        #pragma unroll
        for (int nt = 0; nt < 2; ++nt)
            bf[nt] = loadB_fragT(&lB[cur][(wn * 32 + nt * 16 + l16) * 32], h);
        #pragma unroll
        for (int mt = 0; mt < 4; ++mt)
            #pragma unroll
            for (int nt = 0; nt < 2; ++nt)
                acc[mt][nt] = wmma_bf(af[mt], bf[nt], acc[mt][nt]);
        __syncthreads();
    }

    if (EPI == 0) {
        #pragma unroll
        for (int mt = 0; mt < 4; ++mt)
            #pragma unroll
            for (int nt = 0; nt < 2; ++nt)
                #pragma unroll
                for (int v = 0; v < 8; ++v) {
                    int gm = m0 + wm * 64 + mt * 16 + v + 8 * h;
                    int gn = n0 + wn * 32 + nt * 16 + l16;
                    C[(size_t)gm * ldc + gn] = acc[mt][nt][v];
                }
    } else {
        const float* nm = normBase + (size_t)bz * normBS;
        float local = 0.f;
        for (int mt = 0; mt < 4; ++mt)
            for (int nt = 0; nt < 2; ++nt)
                for (int v = 0; v < 8; ++v) {
                    int gm = m0 + wm * 64 + mt * 16 + v + 8 * h;
                    int gn = n0 + wn * 32 + nt * 16 + l16;
                    float c = acc[mt][nt][v] / (nm[gm] * nm[gn] + 1e-10f);
                    if (gm != gn) local += c * c;
                }
        atomicAdd(&cosacc[bz], local);
    }
}

// ---------------------------------------------------------------------------
// 5. gather G for one batch: [grouped_pts | grouped_xyz | xyz | pts | re | 0pad]
// ---------------------------------------------------------------------------
__global__ __launch_bounds__(256) void k_gather(
    const float* __restrict__ points, const float* __restrict__ xyz,
    const float* __restrict__ re, const int* __restrict__ gidx,
    unsigned short* __restrict__ G, int b) {
    const int n = blockIdx.x, tid = threadIdx.x;
    __shared__ int gi[NS];
    if (tid < NS) gi[tid] = gidx[((size_t)b * NPT + n) * NS + tid];
    __syncthreads();
    unsigned short* row = G + (size_t)n * NGP;
    for (int c = tid; c < NGP; c += 256) {
        float v = 0.f;
        if (c < 2048)      { int k = c >> 6, d = c & 63; v = points[((size_t)b * DIMD + d) * NPT + gi[k]]; }
        else if (c < 2144) { int t = c - 2048; int k = t / 3, cc = t - 3 * k; v = xyz[((size_t)b * 3 + cc) * NPT + gi[k]]; }
        else if (c < 2147) { v = xyz[((size_t)b * 3 + (c - 2144)) * NPT + n]; }
        else if (c < 2211) { v = points[((size_t)b * DIMD + (c - 2147)) * NPT + n]; }
        else if (c < NG)   { v = re[((size_t)b * NPT + n) * 6 + (c - 2211)]; }
        row[c] = f2bf(v);
    }
}

// ---------------------------------------------------------------------------
// 6. weight swizzle: fp32 (Nout x Kin) -> bf16 B-fragments (kt,nt,lane,e)
// ---------------------------------------------------------------------------
__global__ void k_swizzle(const float* __restrict__ w, unsigned short* __restrict__ dst,
                          int Kin, int Ntiles, int total) {
    int i = blockIdx.x * 256 + threadIdx.x;
    if (i >= total) return;
    int e = i & 15, lane = (i >> 4) & 31, tile = i >> 9;
    int kt = tile / Ntiles, nt = tile - kt * Ntiles;
    int k = kt * 32 + e + 16 * (lane >> 4);
    int nn = nt * 16 + (lane & 15);
    dst[i] = f2bf((k < Kin) ? w[nn * Kin + k] : 0.f);
}

// ---------------------------------------------------------------------------
// 7. edge MLP: per (b,s) 32x134 -> 128 -> 128 -> 256, max over 32, WMMA
// ---------------------------------------------------------------------------
static __device__ inline void mlp_layer(const unsigned short* lin, int ldin, int ktc,
                                        const unsigned short* wz, int ntTotal, int ntPerWave,
                                        unsigned short* lout, int ldout,
                                        const float* bias, const float* g, const float* bb,
                                        int wave, int lane) {
    const int h = lane >> 4, l16 = lane & 15;
    const float bns = rsqrtf(1.f + EPSBN);
    for (int t = 0; t < ntPerWave; ++t) {
        const int nt = wave * ntPerWave + t;
        v8f a0 = vzero8(), a1 = vzero8();
        for (int kt = 0; kt < ktc; ++kt) {
            v16bf fa0 = loadA_frag(lin + l16 * ldin, kt * 32, h);
            v16bf fa1 = loadA_frag(lin + (16 + l16) * ldin, kt * 32, h);
            v16bf fb  = loadB_swz(wz + (((kt * ntTotal + nt) * 32 + lane) << 4));
            a0 = wmma_bf(fa0, fb, a0);
            a1 = wmma_bf(fa1, fb, a1);
        }
        const int n = nt * 16 + l16;
        const float sc = g[n] * bns, bc = bb[n], bi = bias[n];
        #pragma unroll
        for (int v = 0; v < 8; ++v) {
            int m = v + 8 * h;
            float x0 = (a0[v] + bi) * sc + bc; x0 = x0 > 0.f ? x0 : 0.2f * x0;
            float x1 = (a1[v] + bi) * sc + bc; x1 = x1 > 0.f ? x1 : 0.2f * x1;
            lout[m * ldout + n]        = f2bf(x0);
            lout[(16 + m) * ldout + n] = f2bf(x1);
        }
    }
}

__global__ __launch_bounds__(128) void k_edge(
    const float* __restrict__ Y, const unsigned short* __restrict__ wz,
    const float* mb0, const float* g0, const float* b0,
    const float* mb1, const float* g1, const float* b1,
    const float* mb2, const float* g2, const float* b2,
    float* __restrict__ outNP) {
    const int tid = threadIdx.x, wave = tid >> 5, lane = tid & 31;
    const int s = blockIdx.x, b = blockIdx.y;
    const float* yr = Y + ((size_t)b * SN + s) * NGP;
    __shared__ unsigned short X [32 * 160];
    __shared__ unsigned short A1[32 * 128];
    __shared__ unsigned short A2[32 * 128];
    for (int i = tid; i < 32 * 160; i += 128) {
        int r = i / 160, k = i - r * 160;
        float v = 0.f;
        if (k < 64)        v = yr[2147 + k];
        else if (k < 128)  { int d = k - 64;  v = yr[r * 64 + d] - yr[2147 + d]; }
        else if (k < 131)  v = yr[2144 + (k - 128)];
        else if (k < 134)  { int c = k - 131; v = yr[2048 + r * 3 + c] - yr[2144 + c]; }
        X[i] = f2bf(v);
    }
    __syncthreads();
    mlp_layer(X, 160, 5, wz,         8, 2, A1, 128, mb0, g0, b0, wave, lane);
    __syncthreads();
    mlp_layer(A1, 128, 4, wz + 20480, 8, 2, A2, 128, mb1, g1, b1, wave, lane);
    __syncthreads();
    // layer 3 (128->256) with fused max-over-samples epilogue
    const int h = lane >> 4, l16 = lane & 15;
    const float bns = rsqrtf(1.f + EPSBN);
    for (int t = 0; t < 4; ++t) {
        const int nt = wave * 4 + t;
        v8f a0 = vzero8(), a1 = vzero8();
        for (int kt = 0; kt < 4; ++kt) {
            v16bf fa0 = loadA_frag(A2 + l16 * 128, kt * 32, h);
            v16bf fa1 = loadA_frag(A2 + (16 + l16) * 128, kt * 32, h);
            v16bf fb  = loadB_swz(wz + 36864 + (((kt * 16 + nt) * 32 + lane) << 4));
            a0 = wmma_bf(fa0, fb, a0);
            a1 = wmma_bf(fa1, fb, a1);
        }
        const int n = nt * 16 + l16;
        const float sc = g2[n] * bns, bc = b2[n], bi = mb2[n];
        float mx = -3.0e38f;
        #pragma unroll
        for (int v = 0; v < 8; ++v) {
            float x0 = (a0[v] + bi) * sc + bc; x0 = x0 > 0.f ? x0 : 0.2f * x0;
            float x1 = (a1[v] + bi) * sc + bc; x1 = x1 > 0.f ? x1 : 0.2f * x1;
            mx = fmaxf(mx, fmaxf(x0, x1));
        }
        mx = fmaxf(mx, __shfl_xor(mx, 16, 32));
        if (h == 0) outNP[((size_t)b * 256 + n) * SN + s] = mx;
    }
}

// ---------------------------------------------------------------------------
// 8. scatter sample outputs from Y; 9. finalize scalars
// ---------------------------------------------------------------------------
#define O1 0u
#define O2 24576u
#define O3 2121728u
#define O4 2121729u
#define O5 2121730u
#define O6 2646018u

__global__ __launch_bounds__(128) void k_scatter(const float* __restrict__ Y,
                                                 float* __restrict__ out) {
    const int s = blockIdx.x, b = blockIdx.y, tid = threadIdx.x;
    const float* yr = Y + ((size_t)b * SN + s) * NGP;
    if (tid < 3)        out[O1 + ((size_t)b * 3 + tid) * SN + s] = yr[2144 + tid];
    else if (tid < 67)  { int d = tid - 3;  out[O5 + ((size_t)b * SN + s) * 64 + d] = yr[2147 + d]; }
    else if (tid < 73)  { int f = tid - 67; out[O6 + ((size_t)b * SN + s) * 6 + f]  = yr[2211 + f]; }
}

__global__ __launch_bounds__(256) void k_finalize(const float* __restrict__ cosacc,
                                                  const int* __restrict__ flags,
                                                  float* __restrict__ out) {
    __shared__ int red[256];
    const int tid = threadIdx.x;
    int c = 0;
    for (int i = tid; i < NPT; i += 256) c += flags[i];
    red[tid] = c; __syncthreads();
    for (int off = 128; off; off >>= 1) { if (tid < off) red[tid] += red[tid + off]; __syncthreads(); }
    if (tid == 0) {
        out[O4] = (float)red[0];
        float cl = 0.f;
        for (int b = 0; b < BQ; ++b) cl += sqrtf(cosacc[b]);
        out[O3] = cl / (float)BQ;
    }
}

// ---------------------------------------------------------------------------
// launch
// ---------------------------------------------------------------------------
extern "C" void kernel_launch(void* const* d_in, const int* in_sizes, int n_in,
                              void* d_out, int out_size, void* d_ws, size_t ws_size,
                              hipStream_t stream) {
    const float* re    = (const float*)d_in[0];
    const float* xyz   = (const float*)d_in[1];
    const float* pts   = (const float*)d_in[2];
    const float* noise = (const float*)d_in[3];
    const float* w1w = (const float*)d_in[4],  *w1b = (const float*)d_in[5];
    const float* bg1 = (const float*)d_in[6],  *bb1 = (const float*)d_in[7];
    const float* w2w = (const float*)d_in[8],  *w2b = (const float*)d_in[9];
    const float* bg2 = (const float*)d_in[10], *bb2 = (const float*)d_in[11];
    const float* mw0 = (const float*)d_in[12], *mb0 = (const float*)d_in[13];
    const float* g0  = (const float*)d_in[14], *b0  = (const float*)d_in[15];
    const float* mw1 = (const float*)d_in[16], *mb1 = (const float*)d_in[17];
    const float* g1  = (const float*)d_in[18], *b1  = (const float*)d_in[19];
    const float* mw2 = (const float*)d_in[20], *mb2 = (const float*)d_in[21];
    const float* g2  = (const float*)d_in[22], *b2  = (const float*)d_in[23];
    float* out = (float*)d_out;

    char* ws = (char*)d_ws;                       // workspace layout (~351 MB)
    int*            gidx   = (int*)           (ws + 0);             //   4.0 MB
    float*          logits = (float*)         (ws + 4194304);       // 128.0 MB
    unsigned short* swb    = (unsigned short*)(ws + 138412032);     //  64.0 MB
    unsigned short* swbT   = (unsigned short*)(ws + 205520896);     //  64.0 MB
    float*          norm   = (float*)         (ws + 272629760);     //  32 KB
    float*          cosacc = (float*)         (ws + 272662528);
    int*            idx0   = (int*)           (ws + 272662784);
    int*            flags  = (int*)           (ws + 272666880);
    unsigned short* wz     = (unsigned short*)(ws + 272683264);     // 136 KB
    unsigned short* G      = (unsigned short*)(ws + 272822528);     //  18.9 MB (per-batch reuse)
    float*          Y      = (float*)         (ws + 291696896);     //  75.5 MB

    k_init<<<17, 256, 0, stream>>>(cosacc, flags);
    k_ballquery<<<dim3(16, BQ), 256, 0, stream>>>(xyz, gidx);
    k_logits<<<dim3(NPT, BQ), 256, 0, stream>>>(pts, xyz, noise, w1w, w1b, bg1, bb1,
                                                w2w, w2b, bg2, bb2, logits);
    k_softmax<<<dim3(SN, BQ), 256, 0, stream>>>(logits, swb, swbT, norm, idx0);
    k_markflags<<<4, 256, 0, stream>>>(idx0, flags);

    k_swizzle<<<(20480 + 255) / 256, 256, 0, stream>>>(mw0, wz,          134, 8,  20480);
    k_swizzle<<<(16384 + 255) / 256, 256, 0, stream>>>(mw1, wz + 20480,  128, 8,  16384);
    k_swizzle<<<(32768 + 255) / 256, 256, 0, stream>>>(mw2, wz + 36864,  128, 16, 32768);

    // inner = sw @ sw^T with fused cos-loss epilogue (M=N=1024, K=4096, per batch)
    k_gemm<1><<<dim3(8, 8, BQ), 256, 0, stream>>>(swb, swbT, nullptr, NPT, SN, 0,
                                                  (size_t)SN * NPT, (size_t)NPT * SN,
                                                  norm, SN, cosacc);
    // Y = sw @ G (M=1024, K=4096, N=2304 padded), per batch with G reuse
    for (int b = 0; b < BQ; ++b) {
        k_gather<<<NPT, 256, 0, stream>>>(pts, xyz, re, gidx, G, b);
        k_gemm<0><<<dim3(18, 8), 256, 0, stream>>>(swb + (size_t)b * SN * NPT, G,
                                                   Y + (size_t)b * SN * NGP, NPT, NGP, NGP,
                                                   0, 0, nullptr, 0, nullptr);
    }

    k_scatter<<<dim3(SN, BQ), 128, 0, stream>>>(Y, out);
    k_edge<<<dim3(SN, BQ), 128, 0, stream>>>(Y, wz, mb0, g0, b0, mb1, g1, b1, mb2, g2, b2,
                                             out + O2);
    k_finalize<<<1, 256, 0, stream>>>(cosacc, flags, out);
    (void)in_sizes; (void)n_in; (void)out_size; (void)ws_size;
}
